// GraphDegreeConv_56710748176712
// MI455X (gfx1250) — matrix operand order repaired and checked by reading, not compiled, not run on hardware
//
#include <hip/hip_runtime.h>

// GraphDegreeConv fused kernel for MI455X (gfx1250, wave32, WMMA).
//
// Pipeline (all on `stream`, deterministic, no atomics):
//   1) gdc_prep_weights : pack [W_d ; W_self] (K=272 -> pad 288) into WMMA
//      B-fragment order as split-precision f16 (hi + lo) in d_ws (L2-resident).
//   2) gdc_fused        : per 16-row tile of a degree group, gather
//      [a_sum | b_sum | own] and split fp32 -> (hi,lo) f16 ONCE into LDS,
//      then 8 waves x (9 ksteps x 3 WMMA) v_wmma_f32_16x16x32_f16 with the
//      Markidis hi/lo decomposition => ~fp32 accuracy at f16 matrix rate.
//      Scatter x rows to d_out (pre-BN). Bias cancels in BN -> dropped.
//   3) colstats_partial / colstats_final : deterministic per-feature
//      sum / sumsq reduction (512 fixed partials).
//   4) bn_relu          : x = relu((x - mean) * rsqrt(var + eps)) in-place.

typedef __attribute__((ext_vector_type(16))) _Float16 v16h;
typedef __attribute__((ext_vector_type(8)))  _Float16 v8h;
typedef __attribute__((ext_vector_type(8)))  float    v8f;

#define NODE    128
#define EDGE    16
#define OUTF    128
#define KTOT    288      // 128 (a_sum) + 16 (b_sum) + 128 (own) + 16 zero-pad
#define KSTEPS  9        // 288 / 32
#define PITCHH  296      // LDS row pitch in halves: 592B = 148 words; m*148 mod 64
                         // = 4*(5m mod 16) -> 16 distinct bank quads, conflict-free
#define NATOMS  500000

// tiles per degree group: ceil(n_d / 16)
#define T1 3125
#define T2 9375
#define T3 12500
#define T4 6250
#define TILES_TOTAL (T1 + T2 + T3 + T4)   // 31250

// ---------------------------------------------------------------------------
// 1) Weight pre-swizzle into WMMA B-fragment order, split f16 hi/lo.
//    B layout for V_WMMA_F32_16X16X32_F16 (K=32 x N=16 per fragment):
//      lane L: n = L&15, K = (L<16 ? 0 : 8) + (e<8 ? e : e+8)  for element e in 0..15.
//    Fragment index: ((g*9 + s)*32 + lane), g = degree*8 + ntile.
// ---------------------------------------------------------------------------
__global__ __launch_bounds__(256) void gdc_prep_weights(
    const float* __restrict__ Wself,
    const float* __restrict__ W1, const float* __restrict__ W2,
    const float* __restrict__ W3, const float* __restrict__ W4,
    _Float16* __restrict__ hi, _Float16* __restrict__ lo)
{
    int gid = blockIdx.x * 256 + threadIdx.x;
    const int total = 4 * 8 * KSTEPS * 32 * 16;   // 147456
    if (gid >= total) return;
    int e    =  gid        & 15;
    int lane = (gid >> 4)  & 31;
    int tmp  =  gid >> 9;
    int s    =  tmp % KSTEPS;
    int g    =  tmp / KSTEPS;
    int dI   =  g >> 3;          // degree index 0..3
    int t    =  g & 7;           // n-tile 0..7
    int n = t * 16 + (lane & 15);
    int k = s * 32 + ((lane < 16) ? 0 : 8) + ((e < 8) ? e : 16 + (e - 8));
    const float* Wd = (dI == 0) ? W1 : (dI == 1) ? W2 : (dI == 2) ? W3 : W4;
    float v = 0.0f;
    if (k < 144)      v = Wd[k * 128 + n];            // rows 0..127 a-part, 128..143 b-part
    else if (k < 272) v = Wself[(k - 144) * 128 + n]; // self weights
    _Float16 h = (_Float16)v;
    hi[gid] = h;
    lo[gid] = (_Float16)(v - (float)h);
}

// ---------------------------------------------------------------------------
// 2) Fused gather + split-precision WMMA GEMM + scatter.
// ---------------------------------------------------------------------------
__device__ __forceinline__ void split_store8(_Float16* __restrict__ dhi,
                                             _Float16* __restrict__ dlo,
                                             const float* __restrict__ v)
{
    v8h h, l;
    #pragma unroll
    for (int c = 0; c < 8; ++c) {
        _Float16 x = (_Float16)v[c];
        h[c] = x;
        l[c] = (_Float16)(v[c] - (float)x);
    }
    *(v8h*)dhi = h;    // 16B LDS store, 16B-aligned (offsets are multiples of 8 halves)
    *(v8h*)dlo = l;
}

__global__ __launch_bounds__(256) void gdc_fused(
    const float* __restrict__ atom, const float* __restrict__ bondr,
    const int* __restrict__ a1, const int* __restrict__ b1,
    const int* __restrict__ a2, const int* __restrict__ b2,
    const int* __restrict__ a3, const int* __restrict__ b3,
    const int* __restrict__ a4, const int* __restrict__ b4,
    const _Float16* __restrict__ wHi, const _Float16* __restrict__ wLo,
    float* __restrict__ xout)
{
    // A tile, pre-split into hi/lo f16: 2 x 16 x 296 halves = 18.5 KB
    __shared__ _Float16 AsHi[16 * PITCHH];
    __shared__ _Float16 AsLo[16 * PITCHH];

    int bt = blockIdx.x;
    int dI, tileLocal, nd;
    const int* aidx; const int* bidx;
    if (bt < T1)                 { dI = 0; tileLocal = bt;                  nd = 50000;  aidx = a1; bidx = b1; }
    else if (bt < T1 + T2)       { dI = 1; tileLocal = bt - T1;             nd = 150000; aidx = a2; bidx = b2; }
    else if (bt < T1 + T2 + T3)  { dI = 2; tileLocal = bt - (T1 + T2);      nd = 200000; aidx = a3; bidx = b3; }
    else                         { dI = 3; tileLocal = bt - (T1 + T2 + T3); nd = 100000; aidx = a4; bidx = b4; }
    const int d = dI + 1;

    const int tid = threadIdx.x;

    // ---- cooperative gather + one-time fp32 -> (hi,lo) split into LDS
    {
        int m = tid >> 4;        // row in tile 0..15
        int j = tid & 15;        // 8-float column chunk of the 128-wide parts
        int r = tileLocal * 16 + m;
        bool valid = r < nd;
        float accA[8] = {0.f,0.f,0.f,0.f,0.f,0.f,0.f,0.f};
        float own [8] = {0.f,0.f,0.f,0.f,0.f,0.f,0.f,0.f};
        if (valid) {
            long ib = (long)r * (d + 1);
            for (int i = 0; i <= d; ++i) {
                int id = aidx[ib + i];
                const float4* p = (const float4*)(atom + (long)id * NODE + j * 8);
                float4 u0 = p[0], u1 = p[1];
                accA[0] += u0.x; accA[1] += u0.y; accA[2] += u0.z; accA[3] += u0.w;
                accA[4] += u1.x; accA[5] += u1.y; accA[6] += u1.z; accA[7] += u1.w;
                if (i == 0) {
                    own[0] = u0.x; own[1] = u0.y; own[2] = u0.z; own[3] = u0.w;
                    own[4] = u1.x; own[5] = u1.y; own[6] = u1.z; own[7] = u1.w;
                }
            }
        }
        _Float16* rH = &AsHi[m * PITCHH];
        _Float16* rL = &AsLo[m * PITCHH];
        split_store8(rH +       j * 8, rL +       j * 8, accA);  // a_sum   [0..127]
        split_store8(rH + 144 + j * 8, rL + 144 + j * 8, own);   // own     [144..271]
        if (j < 2) {   // b_sum: 16 floats per row -> cols [128..143]
            float accB[8] = {0.f,0.f,0.f,0.f,0.f,0.f,0.f,0.f};
            if (valid) {
                long jb = (long)r * d;
                for (int i = 0; i < d; ++i) {
                    int id = bidx[jb + i];
                    const float4* p = (const float4*)(bondr + (long)id * EDGE + j * 8);
                    float4 u0 = p[0], u1 = p[1];
                    accB[0] += u0.x; accB[1] += u0.y; accB[2] += u0.z; accB[3] += u0.w;
                    accB[4] += u1.x; accB[5] += u1.y; accB[6] += u1.z; accB[7] += u1.w;
                }
            }
            split_store8(rH + 128 + j * 8, rL + 128 + j * 8, accB);
        }
        if (j == 2 || j == 3) {  // zero K-padding [272..287]
            v8h z = {};
            *(v8h*)(rH + 272 + (j - 2) * 8) = z;
            *(v8h*)(rL + 272 + (j - 2) * 8) = z;
        }
    }
    __syncthreads();

    // ---- 8 waves, wave t owns output columns [t*16, t*16+16).
    //      Inner loop is pure WMMA + b128 loads: no per-wave conversion VALU.
    const int lane  = tid & 31;
    const int t     = tid >> 5;
    const int g     = dI * 8 + t;
    const int mA    = lane & 15;
    const int baseA = (lane < 16) ? 0 : 8;   // A-fragment K base per half-wave
    const v16h* WH  = (const v16h*)wHi;
    const v16h* WL  = (const v16h*)wLo;

    v8f acc = {0.f,0.f,0.f,0.f,0.f,0.f,0.f,0.f};

    #pragma unroll
    for (int s = 0; s < KSTEPS; ++s) {
        // A fragment: lane L holds row m=L&15; elements 0..7 = K s*32+base+[0..7],
        // elements 8..15 = K s*32+16+base+[0..7]  -> two ds_load_b128 each for hi/lo.
        const _Float16* pH = &AsHi[mA * PITCHH + s * 32 + baseA];
        const _Float16* pL = &AsLo[mA * PITCHH + s * 32 + baseA];
        v8h h0 = *(const v8h*)(pH);
        v8h h1 = *(const v8h*)(pH + 16);
        v8h l0 = *(const v8h*)(pL);
        v8h l1 = *(const v8h*)(pL + 16);
        v16h ah = __builtin_shufflevector(h0, h1, 0,1,2,3,4,5,6,7,8,9,10,11,12,13,14,15);
        v16h al = __builtin_shufflevector(l0, l1, 0,1,2,3,4,5,6,7,8,9,10,11,12,13,14,15);
        int fi = (g * KSTEPS + s) * 32 + lane;
        v16h bh = WH[fi];
        v16h bl = WL[fi];
        // D = A*B + C, fp32 accumulate; split-precision: hi*hi + lo*hi + hi*lo
        acc = __builtin_amdgcn_wmma_f32_16x16x32_f16(false, ah, false, bh, (short)0, acc, false, false);
        acc = __builtin_amdgcn_wmma_f32_16x16x32_f16(false, al, false, bh, (short)0, acc, false, false);
        acc = __builtin_amdgcn_wmma_f32_16x16x32_f16(false, ah, false, bl, (short)0, acc, false, false);
    }

    // ---- scatter: C/D layout: VGPR r -> (M = r + (lane<16?0:8), N = lane&15)
    const int nCol  = t * 16 + (lane & 15);
    const int rbase = tileLocal * 16 + ((lane < 16) ? 0 : 8);
    #pragma unroll
    for (int rr = 0; rr < 8; ++rr) {
        int grow = rbase + rr;
        if (grow < nd) {
            int node = aidx[(long)grow * (d + 1)];   // own id = first column
            xout[(long)node * OUTF + nCol] = acc[rr];
        }
    }
}

// ---------------------------------------------------------------------------
// 3) Deterministic column statistics (sum, sumsq) in two fixed stages.
// ---------------------------------------------------------------------------
__global__ __launch_bounds__(128) void colstats_partial(
    const float* __restrict__ x, float* __restrict__ psum, float* __restrict__ psq,
    int nRows, int rowsPerBlock)
{
    int f = threadIdx.x;                 // feature 0..127; 128 threads read one row coalesced
    long r0 = (long)blockIdx.x * rowsPerBlock;
    long r1 = r0 + rowsPerBlock; if (r1 > nRows) r1 = nRows;
    float s = 0.f, q = 0.f;
    for (long r = r0; r < r1; ++r) {
        float v = x[r * OUTF + f];
        s += v; q += v * v;
    }
    psum[blockIdx.x * OUTF + f] = s;
    psq [blockIdx.x * OUTF + f] = q;
}

__global__ __launch_bounds__(128) void colstats_final(
    const float* __restrict__ psum, const float* __restrict__ psq,
    float* __restrict__ meanInv, int nPart, float invN)
{
    int f = threadIdx.x;
    float s = 0.f, q = 0.f;
    for (int b = 0; b < nPart; ++b) { s += psum[b * OUTF + f]; q += psq[b * OUTF + f]; }
    float mean = s * invN;
    float var  = q * invN - mean * mean;
    meanInv[f]        = mean;
    meanInv[OUTF + f] = rsqrtf(var + 1e-5f);
}

// ---------------------------------------------------------------------------
// 4) In-place BatchNorm (affine=False, biased stats) + ReLU.
// ---------------------------------------------------------------------------
__global__ __launch_bounds__(256) void bn_relu(
    float* __restrict__ x, const float* __restrict__ meanInv, long total)
{
    long i = (long)blockIdx.x * blockDim.x + threadIdx.x;
    long stride = (long)gridDim.x * blockDim.x;
    for (; i < total; i += stride) {
        int f = (int)(i & (OUTF - 1));
        float v = (x[i] - meanInv[f]) * meanInv[OUTF + f];
        x[i] = v > 0.f ? v : 0.f;
    }
}

// ---------------------------------------------------------------------------
extern "C" void kernel_launch(void* const* d_in, const int* in_sizes, int n_in,
                              void* d_out, int out_size, void* d_ws, size_t ws_size,
                              hipStream_t stream) {
    const float* atom  = (const float*)d_in[0];
    const float* bondr = (const float*)d_in[1];
    const int*   a1 = (const int*)d_in[2];
    const int*   b1 = (const int*)d_in[3];
    const int*   a2 = (const int*)d_in[4];
    const int*   b2 = (const int*)d_in[5];
    const int*   a3 = (const int*)d_in[6];
    const int*   b3 = (const int*)d_in[7];
    const int*   a4 = (const int*)d_in[8];
    const int*   b4 = (const int*)d_in[9];
    const float* Wself = (const float*)d_in[10];
    const float* W1 = (const float*)d_in[11];
    const float* W2 = (const float*)d_in[12];
    const float* W3 = (const float*)d_in[13];
    const float* W4 = (const float*)d_in[14];
    // d_in[15] = bias: constant per feature, cancels exactly under BatchNorm.
    float* out = (float*)d_out;

    // Workspace layout (~1.12 MB total)
    char* ws = (char*)d_ws;
    const size_t WPACK = 4u * 8u * KSTEPS * 32u * 16u;   // 147456 halfs per buffer
    _Float16* wHi    = (_Float16*)(ws);
    _Float16* wLo    = (_Float16*)(ws + WPACK * 2);                    // +294912
    float*    psum   = (float*)   (ws + WPACK * 4);                    // +589824
    float*    psq    = (float*)   (ws + WPACK * 4 + 512 * OUTF * 4);   // +262144
    float*    meanIv = (float*)   (ws + WPACK * 4 + 2 * 512 * OUTF * 4);

    gdc_prep_weights<<<(int)((WPACK + 255) / 256), 256, 0, stream>>>(
        Wself, W1, W2, W3, W4, wHi, wLo);

    gdc_fused<<<TILES_TOTAL, 256, 0, stream>>>(
        atom, bondr, a1, b1, a2, b2, a3, b3, a4, b4, wHi, wLo, out);

    colstats_partial<<<512, 128, 0, stream>>>(out, psum, psq, NATOMS, 977);
    colstats_final<<<1, 128, 0, stream>>>(psum, psq, meanIv, 512, 1.0f / (float)NATOMS);
    bn_relu<<<2048, 256, 0, stream>>>(out, meanIv, (long)NATOMS * OUTF);
}
